// GNN_20968030339503
// MI455X (gfx1250) — matrix operand order repaired and checked by reading, not compiled, not run on hardware
//
#include <hip/hip_runtime.h>
#include <hip/hip_bf16.h>
#include <math.h>

// ---------------- problem constants ----------------
#define NP 100000
#define NA 50000
#define HD 128
#define EC 1000000
#define EW 320000
// padded row counts (multiples of 128 so each 256-thread GEMM block = 8 wave tiles)
#define NPp 100096
#define NAp 50048

typedef __attribute__((ext_vector_type(16))) _Float16 v16h;
typedef __attribute__((ext_vector_type(8)))  _Float16 v8h;
typedef __attribute__((ext_vector_type(8)))  float    v8f;

// ---------------- f32 -> f16 convert (zero-pads tail rows) ----------------
__global__ void k_cvt_f16(const float* __restrict__ src, _Float16* __restrict__ dst,
                          long nvalid, long ntot) {
    long i = (long)blockIdx.x * blockDim.x + threadIdx.x;
    if (i < ntot) dst[i] = (i < nvalid) ? (_Float16)src[i] : (_Float16)0.0f;
}

// ---------------- edge scatter-add: sum[dst] += x[src], cnt[dst] += 1 ----------------
// 32 threads per edge, 4 floats each (float4), atomics land in L2 (sum fits in 192MB L2)
__global__ void k_scatter(const float* __restrict__ xsrc,
                          const int* __restrict__ src, const int* __restrict__ dst,
                          float* __restrict__ sum, float* __restrict__ cnt, long nedges) {
    long gid = (long)blockIdx.x * blockDim.x + threadIdx.x;
    long e = gid >> 5;
    if (e >= nedges) return;
    int j = (int)(gid & 31) * 4;
    int s = src[e], d = dst[e];
    const float4 v = *(const float4*)(xsrc + (long)s * HD + j);
    float* o = sum + (long)d * HD + j;
    atomicAdd(o + 0, v.x);
    atomicAdd(o + 1, v.y);
    atomicAdd(o + 2, v.z);
    atomicAdd(o + 3, v.w);
    if (j == 0) atomicAdd(cnt + d, 1.0f);
}

// ---------------- mean = sum / max(cnt,1), convert to f16 ----------------
__global__ void k_mean_cvt(const float* __restrict__ sum, const float* __restrict__ cnt,
                           _Float16* __restrict__ out, long ntot) {
    long i = (long)blockIdx.x * blockDim.x + threadIdx.x;
    if (i >= ntot) return;
    float c = cnt[i >> 7];
    out[i] = (_Float16)(sum[i] / fmaxf(c, 1.0f));
}

// ---------------- WMMA GEMM: Out = Amean@Wl + Axdst@Wr + bias ----------------
// One wave32 owns a full 16x128 output tile (8 col-tiles of v_wmma_f32_16x16x32_f16).
// Weights staged transposed in LDS (Wt[n][k]) so each B fragment is two runs of 8 contiguous halves.
__global__ __launch_bounds__(256)
void sage_gemm(const _Float16* __restrict__ Amean,
               const _Float16* __restrict__ Axdst,
               const _Float16* __restrict__ Wl,   // [K=128][N=128] f16 row-major
               const _Float16* __restrict__ Wr,
               const float*    __restrict__ bias, // [128] f32
               float* __restrict__ Out,           // [Mp][128] f32
               int Mp) {
    __shared__ _Float16 ldsWt[128 * 128];  // 32 KB, transposed [n][k]
    const int tid  = threadIdx.x;
    const int wid  = tid >> 5;
    const int lane = tid & 31;
    const int half = lane >> 4;   // 0 or 1
    const int nloc = lane & 15;   // column within 16-wide tile
    const long m0  = ((long)blockIdx.x * 8 + wid) * 16;

    v8f acc[8];
#pragma unroll
    for (int i = 0; i < 8; ++i) acc[i] = (v8f){0.f, 0.f, 0.f, 0.f, 0.f, 0.f, 0.f, 0.f};

#pragma unroll
    for (int phase = 0; phase < 2; ++phase) {
        const _Float16* Wsrc = phase ? Wr : Wl;
        const _Float16* Ab   = phase ? Axdst : Amean;
        __syncthreads();  // all waves done with previous LDS contents
        for (int idx = tid; idx < 128 * 128; idx += 256) {
            int k = idx >> 7, n = idx & 127;
            ldsWt[n * 128 + k] = Wsrc[idx];  // transpose into LDS
        }
        __syncthreads();
#pragma unroll
        for (int ks = 0; ks < 4; ++ks) {
            const int kbase = ks * 32;
            // A 16x32 f16 fragment (ISA layout): lane row m = nloc;
            // e<8 -> K = kbase + half*8 + e ; e>=8 -> K = kbase + 16 + half*8 + (e-8)
            const _Float16* arow = Ab + (m0 + nloc) * HD + kbase + half * 8;
            v8h lo = *(const v8h*)(arow);
            v8h hi = *(const v8h*)(arow + 16);
            v16h a = __builtin_shufflevector(lo, hi, 0, 1, 2, 3, 4, 5, 6, 7,
                                             8, 9, 10, 11, 12, 13, 14, 15);
#pragma unroll
            for (int nt = 0; nt < 8; ++nt) {
                // B 32x16 fragment: lane holds column (nt*16+nloc); e<8 -> K = kbase + half*8 + e ; e>=8 -> K = kbase + 16 + half*8 + (e-8)
                const _Float16* bp = ldsWt + (nt * 16 + nloc) * 128 + kbase + half * 8;
                v8h b0 = *(const v8h*)(bp);
                v8h b1 = *(const v8h*)(bp + 16);
                v16h b = __builtin_shufflevector(b0, b1, 0, 1, 2, 3, 4, 5, 6, 7,
                                                 8, 9, 10, 11, 12, 13, 14, 15);
                acc[nt] = __builtin_amdgcn_wmma_f32_16x16x32_f16(
                    false, a, false, b, (short)0, acc[nt], false, false);
            }
        }
    }

    // epilogue: D layout — acc[nt][r] = Out[m0 + r + 8*half][nt*16 + nloc]
#pragma unroll
    for (int nt = 0; nt < 8; ++nt) {
        float bv = bias[nt * 16 + nloc];
#pragma unroll
        for (int r = 0; r < 8; ++r) {
            long m = m0 + r + 8 * half;
            Out[m * HD + nt * 16 + nloc] = acc[nt][r] + bv;
        }
    }
}

// ---------------- row L2-normalize + scale + (optional accumulate) + (optional relu) ----
// one 128-thread block per valid row
__global__ void k_norm(const float* __restrict__ src, float* __restrict__ dst,
                       float scale, int add_prev, int do_relu) {
    long row = blockIdx.x;
    int t = threadIdx.x;
    long idx = row * HD + t;
    float v = src[idx];
    float ss = v * v;
#pragma unroll
    for (int off = 16; off > 0; off >>= 1) ss += __shfl_xor(ss, off, 32);
    __shared__ float red[4];
    if ((t & 31) == 0) red[t >> 5] = ss;
    __syncthreads();
    float nrm = sqrtf(red[0] + red[1] + red[2] + red[3]);
    float o = v / fmaxf(nrm, 1e-12f) * scale;
    if (add_prev) o += dst[idx];
    if (do_relu) o = fmaxf(o, 0.0f);
    dst[idx] = o;
}

// ---------------- one SAGE conv (aggregate + gemm + normalize/combine) ----------------
static void run_conv(const float* xsrc, const int* esrc, const int* edst, long nedges,
                     const _Float16* xdst_h, long Mp,
                     const float* Wl, const float* bl, const float* Wr,
                     _Float16* wl_h, _Float16* wr_h, _Float16* mean_h,
                     float* sumbuf /* aliases GEMM Out */, float* cntbuf,
                     float* out_dst, long Mvalid,
                     float scale, int add_prev, int do_relu, hipStream_t stream) {
    hipMemsetAsync(sumbuf, 0, (size_t)Mp * HD * sizeof(float), stream);
    hipMemsetAsync(cntbuf, 0, (size_t)Mp * sizeof(float), stream);
    long nth = nedges * 32;
    k_scatter<<<(int)((nth + 255) / 256), 256, 0, stream>>>(xsrc, esrc, edst, sumbuf, cntbuf, nedges);
    long ntot = Mp * HD;
    k_mean_cvt<<<(int)((ntot + 255) / 256), 256, 0, stream>>>(sumbuf, cntbuf, mean_h, ntot);
    k_cvt_f16<<<64, 256, 0, stream>>>(Wl, wl_h, HD * HD, HD * HD);
    k_cvt_f16<<<64, 256, 0, stream>>>(Wr, wr_h, HD * HD, HD * HD);
    sage_gemm<<<(int)(Mp / 128), 256, 0, stream>>>(mean_h, xdst_h, wl_h, wr_h, bl, sumbuf, (int)Mp);
    k_norm<<<(int)Mvalid, 128, 0, stream>>>(sumbuf, out_dst, scale, add_prev, do_relu);
}

extern "C" void kernel_launch(void* const* d_in, const int* in_sizes, int n_in,
                              void* d_out, int out_size, void* d_ws, size_t ws_size,
                              hipStream_t stream) {
    const float* xp = (const float*)d_in[0];
    const float* xa = (const float*)d_in[1];
    const int* cs = (const int*)d_in[2];
    const int* cd = (const int*)d_in[3];
    const int* wsrc = (const int*)d_in[4];
    const int* wdst = (const int*)d_in[5];
    const int* rs = (const int*)d_in[6];
    const int* rd = (const int*)d_in[7];
    // weight order per layer/edge-type: Wl, bl, Wr
    const float* P[18];
    for (int i = 0; i < 18; ++i) P[i] = (const float*)d_in[8 + i];
    // l0: cites P[0..2], writes P[3..5], rev P[6..8]; l1: P[9..17]

    // -------- workspace partition (256B aligned) --------
    char* wp = (char*)d_ws;
    auto take = [&](size_t bytes) { char* r = wp; wp += (bytes + 255) & ~(size_t)255; return r; };
    float*    sumbuf  = (float*)take((size_t)NPp * HD * 4);  // also GEMM output (time-disjoint)
    float*    cntbuf  = (float*)take((size_t)NPp * 4);
    _Float16* mean_h  = (_Float16*)take((size_t)NPp * HD * 2);
    _Float16* xdstp_h = (_Float16*)take((size_t)NPp * HD * 2);
    _Float16* xdsta_h = (_Float16*)take((size_t)NAp * HD * 2);
    float*    nxp     = (float*)take((size_t)NPp * HD * 4);
    float*    nxa     = (float*)take((size_t)NAp * HD * 4);
    _Float16* wl_h    = (_Float16*)take((size_t)HD * HD * 2);
    _Float16* wr_h    = (_Float16*)take((size_t)HD * HD * 2);

    const long NPf = (long)NP * HD, NAf = (long)NA * HD;
    const long NPpf = (long)NPp * HD, NApf = (long)NAp * HD;

    // ================= layer 0 =================
    k_cvt_f16<<<(int)((NPpf + 255) / 256), 256, 0, stream>>>(xp, xdstp_h, NPf, NPpf);
    k_cvt_f16<<<(int)((NApf + 255) / 256), 256, 0, stream>>>(xa, xdsta_h, NAf, NApf);
    // paper <- cites (src paper), then <- writes (src author); combined 0.5*(a+b), relu at end
    run_conv(xp, cs, cd, EC, xdstp_h, NPp, P[0], P[1], P[2],
             wl_h, wr_h, mean_h, sumbuf, cntbuf, nxp, NP, 0.5f, 0, 0, stream);
    run_conv(xa, wsrc, wdst, EW, xdstp_h, NPp, P[3], P[4], P[5],
             wl_h, wr_h, mean_h, sumbuf, cntbuf, nxp, NP, 0.5f, 1, 1, stream);
    // author <- rev (src paper)
    run_conv(xp, rs, rd, EW, xdsta_h, NAp, P[6], P[7], P[8],
             wl_h, wr_h, mean_h, sumbuf, cntbuf, nxa, NA, 1.0f, 0, 1, stream);

    // ================= layer 1 =================
    k_cvt_f16<<<(int)((NPpf + 255) / 256), 256, 0, stream>>>(nxp, xdstp_h, NPf, NPpf);
    k_cvt_f16<<<(int)((NApf + 255) / 256), 256, 0, stream>>>(nxa, xdsta_h, NAf, NApf);
    float* outp = (float*)d_out;
    float* outa = outp + NPf;
    run_conv(nxp, cs, cd, EC, xdstp_h, NPp, P[9], P[10], P[11],
             wl_h, wr_h, mean_h, sumbuf, cntbuf, outp, NP, 0.5f, 0, 0, stream);
    run_conv(nxa, wsrc, wdst, EW, xdstp_h, NPp, P[12], P[13], P[14],
             wl_h, wr_h, mean_h, sumbuf, cntbuf, outp, NP, 0.5f, 1, 1, stream);
    run_conv(nxp, rs, rd, EW, xdsta_h, NAp, P[15], P[16], P[17],
             wl_h, wr_h, mean_h, sumbuf, cntbuf, outa, NA, 1.0f, 0, 1, stream);
}
